// TransformerAttention_32933809225955
// MI455X (gfx1250) — matrix operand order repaired and checked by reading, not compiled, not run on hardware
//
#include <hip/hip_runtime.h>

typedef __attribute__((ext_vector_type(16))) _Float16 v16h;
typedef __attribute__((ext_vector_type(8)))  _Float16 v8h;
typedef __attribute__((ext_vector_type(8)))  float    v8f;

#define B_  16
#define S_  2048
#define D_  64

union F16x16 { v16h v; _Float16 h[16]; v8h q2[2]; };

static __device__ __forceinline__ v8f wmma16(v16h a, v16h b, v8f c) {
  // D = A(16x32 f16) x B(32x16 f16) + C(16x16 f32)
  return __builtin_amdgcn_wmma_f32_16x16x32_f16(false, a, false, b, (short)0, c, false, false);
}

static __device__ __forceinline__ void cvt8(_Float16* d, float4 a, float4 b) {
  d[0] = (_Float16)a.x; d[1] = (_Float16)a.y; d[2] = (_Float16)a.z; d[3] = (_Float16)a.w;
  d[4] = (_Float16)b.x; d[5] = (_Float16)b.y; d[6] = (_Float16)b.z; d[7] = (_Float16)b.w;
}

// gfx1250 async memory->LDS copy (ASYNCcnt-tracked, no VGPR round trip).
// VDST = LDS byte offset, VADDR = 64-bit global address, saddr = off.
static __device__ __forceinline__ void async_g2l_b64(unsigned lds_off,
                                                     unsigned long long gaddr) {
  asm volatile("global_load_async_to_lds_b64 %0, %1, off"
               :: "v"(lds_off), "v"(gaddr) : "memory");
}
static __device__ __forceinline__ void wait_async0() {
  asm volatile("s_wait_asynccnt 0" ::: "memory");
}
static __device__ __forceinline__ unsigned lds_offset_of(const void* p) {
  return (unsigned)(unsigned long long)(__attribute__((address_space(3))) const void*)p;
}

// ---------------------------------------------------------------------------
// Kernel 1: QKV projection with WMMA.
//   q (scaled by 1/sqrt(D)) -> qh [B][S][D] f16   (row major: A-frag friendly)
//   k                       -> kh [B][S][D] f16   (row major: B-frag for scores)
//   v                       -> vT [B][D][S] f16   (transposed: B-frag for attn@V)
// One wave per 16-row tile of the flattened [B*S] rows; 8 waves per block.
// ---------------------------------------------------------------------------
__global__ void __launch_bounds__(256) qkv_wmma_kernel(
    const float* __restrict__ x,
    const float* __restrict__ Wq, const float* __restrict__ bq,
    const float* __restrict__ Wk, const float* __restrict__ bk,
    const float* __restrict__ Wv, const float* __restrict__ bv,
    _Float16* __restrict__ qh, _Float16* __restrict__ kh, _Float16* __restrict__ vT)
{
  // Weights staged transposed as f16: Whs[p][d*64 + i] = W_p[i][d]
  __shared__ __align__(32) _Float16 Whs[3][64 * 64];
  __shared__ float bs[3][64];

  const int tid = threadIdx.x;
  {
    const float* Wsrcs[3] = {Wq, Wk, Wv};
    const float* bsrcs[3] = {bq, bk, bv};
#pragma unroll
    for (int p = 0; p < 3; ++p) {            // uniform per-projection loops: no
      const float* __restrict__ Wsrc = Wsrcs[p];  // per-lane pointer selects
      for (int i = tid; i < 64 * 64; i += 256) {
        const int r = i >> 6, d = i & 63;
        Whs[p][d * 64 + r] = (_Float16)Wsrc[r * 64 + d];
      }
      if (tid < 64) bs[p][tid] = bsrcs[p][tid];
    }
  }
  __syncthreads();

  const int wave = tid >> 5, lane = tid & 31;
  const int h = lane >> 4, n = lane & 15;          // h: lane half, n: M (A) / N (B,C)
  const int tile = blockIdx.x * 8 + wave;          // 2048 tiles total
  const int row0 = tile * 16;
  const int b    = row0 / S_;

  // A fragments: rows of x, f16. halves[i]=K(f*32+16h+i)
  F16x16 afr[2];
  {
    const float* xr = x + (size_t)(row0 + n) * D_;
#pragma unroll
    for (int f = 0; f < 2; ++f) {
      const float4* g0 = (const float4*)(xr + f * 32 + 16 * h);
      const float4* g1 = (const float4*)(xr + f * 32 + 16 * h + 8);
      cvt8(&afr[f].h[0], g0[0], g0[1]);
      cvt8(&afr[f].h[8], g1[0], g1[1]);
    }
  }

#pragma unroll
  for (int p = 0; p < 3; ++p) {
#pragma unroll
    for (int dt = 0; dt < 4; ++dt) {
      F16x16 b0, b1;  // B frag: halves[i] = W[K = f*32+16h+i][dt*16+n]
      b0.v = *(const v16h*)&Whs[p][(dt * 16 + n) * 64 + 16 * h];
      b1.v = *(const v16h*)&Whs[p][(dt * 16 + n) * 64 + 32 + 16 * h];
      v8f c = {};
      c = wmma16(afr[0].v, b0.v, c);
      c = wmma16(afr[1].v, b1.v, c);
      const float bias = bs[p][dt * 16 + n];
#pragma unroll
      for (int r = 0; r < 8; ++r) {
        const int row = row0 + r + 8 * h;          // C layout: M = r + 8*h
        const int d   = dt * 16 + n;               // N = n
        const float val = c[r] + bias;
        if (p == 0)      qh[(size_t)row * D_ + d] = (_Float16)(val * 0.125f); // fold 1/sqrt(64)
        else if (p == 1) kh[(size_t)row * D_ + d] = (_Float16)val;
        else             vT[((size_t)b * D_ + d) * S_ + (row - b * S_)] = (_Float16)val;
      }
    }
  }
}

// ---------------------------------------------------------------------------
// Kernel 2: initialize d_out[b] = bfc (attention kernel atomically accumulates).
// ---------------------------------------------------------------------------
__global__ void init_out_kernel(float* __restrict__ out, const float* __restrict__ bfc) {
  if (threadIdx.x < B_) out[threadIdx.x] = bfc[0];
}

// ---------------------------------------------------------------------------
// Kernel 3: fused attention (query-axis softmax) + FC dot.
// grid = (8 key-chunks, 16 batches), 256 threads = 8 waves, wave owns 32 keys.
// Pass 1: online column (per-key) max/sum over all 2048 queries via WMMA tiles.
// Pass 2: recompute scores, P = exp(s-m)/Z, bounce P through LDS into A-frag
//         layout, WMMA against V fragments, fold Wfc dot into a wave scalar.
// Q tiles are staged with gfx1250 async global->LDS copies + prefetch.
// ---------------------------------------------------------------------------
__global__ void __launch_bounds__(256) attn_fused_kernel(
    const _Float16* __restrict__ qh, const _Float16* __restrict__ kh,
    const _Float16* __restrict__ vT, const float* __restrict__ Wfc,
    float* __restrict__ out)
{
  __shared__ __align__(16) _Float16 qs[16 * 64];      // shared q tile (2 KB)
  __shared__ __align__(16) _Float16 ps[8][16][32];    // per-wave P bounce (8 KB)

  const int tid  = threadIdx.x;
  const int wave = tid >> 5, lane = tid & 31;
  const int h = lane >> 4, n = lane & 15;
  const int b = blockIdx.y;
  const int keybase = blockIdx.x * 256 + wave * 32;   // this wave's 32 keys

  // B fragments for scores: halves[i] = k[key = keybase+t*16+n][K = f*32+16h+i]
  v16h bk[2][2];
#pragma unroll
  for (int t = 0; t < 2; ++t)
#pragma unroll
    for (int f = 0; f < 2; ++f)
      bk[t][f] = *(const v16h*)(kh + ((size_t)b * S_ + keybase + t * 16 + n) * D_ + f * 32 + 16 * h);

  // B fragments for attn@V: halves[i] = v[key = keybase+16h+i][d = dt*16+n]
  v16h bv[4];
#pragma unroll
  for (int dt = 0; dt < 4; ++dt)
    bv[dt] = *(const v16h*)(vT + ((size_t)b * D_ + dt * 16 + n) * S_ + keybase + 16 * h);

  float mcol[2] = {-3.0e38f, -3.0e38f};  // per-key running max (key = keybase+t*16+n)
  float scol[2] = {0.f, 0.f};            // per-key running sum of exp

  const unsigned long long qbase = (unsigned long long)(qh + (size_t)b * S_ * D_);
  const unsigned qs_lds = lds_offset_of(qs) + (unsigned)tid * 8u;

  // ---------------- pass 1: column softmax statistics ----------------
  for (int qt = 0; qt < S_ / 16; ++qt) {
    __syncthreads();
    async_g2l_b64(qs_lds, qbase + (unsigned long long)qt * 2048ull + (unsigned)tid * 8u);
    __builtin_prefetch((const void*)(qbase + (unsigned long long)(qt + 1) * 2048ull +
                                     (unsigned)tid * 8u), 0, 1);
    wait_async0();
    __syncthreads();
    F16x16 a[2];
#pragma unroll
    for (int f = 0; f < 2; ++f) {
      a[f].q2[0] = *(const v8h*)&qs[n * 64 + f * 32 + 16 * h];
      a[f].q2[1] = *(const v8h*)&qs[n * 64 + f * 32 + 16 * h + 8];
    }
#pragma unroll
    for (int t = 0; t < 2; ++t) {
      v8f c = {};
      c = wmma16(a[0].v, bk[t][0], c);
      c = wmma16(a[1].v, bk[t][1], c);
      float tm = c[0];
#pragma unroll
      for (int r = 1; r < 8; ++r) tm = fmaxf(tm, c[r]);
      tm = fmaxf(tm, __shfl_xor(tm, 16, 32));         // combine both lane halves (rows 0-7 / 8-15)
      const float mnew = fmaxf(mcol[t], tm);
      float psum = 0.f;
#pragma unroll
      for (int r = 0; r < 8; ++r) psum += __expf(c[r] - mnew);
      psum += __shfl_xor(psum, 16, 32);
      scol[t] = scol[t] * __expf(mcol[t] - mnew) + psum;
      mcol[t] = mnew;
    }
  }
  const float inv[2] = {1.f / scol[0], 1.f / scol[1]};

  // ---------------- pass 2: P = softmax column, out += P @ V, fused FC ----------------
  float fcacc = 0.f;
  for (int qt = 0; qt < S_ / 16; ++qt) {
    __syncthreads();
    async_g2l_b64(qs_lds, qbase + (unsigned long long)qt * 2048ull + (unsigned)tid * 8u);
    __builtin_prefetch((const void*)(qbase + (unsigned long long)(qt + 1) * 2048ull +
                                     (unsigned)tid * 8u), 0, 1);
    wait_async0();
    __syncthreads();
    F16x16 a[2];
#pragma unroll
    for (int f = 0; f < 2; ++f) {
      a[f].q2[0] = *(const v8h*)&qs[n * 64 + f * 32 + 16 * h];
      a[f].q2[1] = *(const v8h*)&qs[n * 64 + f * 32 + 16 * h + 8];
    }
#pragma unroll
    for (int t = 0; t < 2; ++t) {
      v8f c = {};
      c = wmma16(a[0].v, bk[t][0], c);
      c = wmma16(a[1].v, bk[t][1], c);
#pragma unroll
      for (int r = 0; r < 8; ++r)
        ps[wave][r + 8 * h][t * 16 + n] = (_Float16)(__expf(c[r] - mcol[t]) * inv[t]);
    }
    __syncthreads();   // P visible; also keeps waves lockstep on qs reuse

    F16x16 ap;         // A frag over local keys: halves[i]=K 16h+i
    ap.q2[0] = *(const v8h*)&ps[wave][n][16 * h];
    ap.q2[1] = *(const v8h*)&ps[wave][n][16 * h + 8];

#pragma unroll
    for (int dt = 0; dt < 4; ++dt) {
      v8f c = {};
      c = wmma16(ap.v, bv[dt], c);
      // out[b][q][d] partial; only consumer is FC dot -> fold Wfc here.
#pragma unroll
      for (int r = 0; r < 8; ++r) {
        const int q = qt * 16 + r + 8 * h;
        const int d = dt * 16 + n;
        fcacc += c[r] * Wfc[(size_t)q * D_ + d];
      }
    }
  }

  // wave reduction of the FC partial, one atomic per wave
#pragma unroll
  for (int off = 16; off > 0; off >>= 1) fcacc += __shfl_xor(fcacc, off, 32);
  if (lane == 0) atomicAdd(&out[b], fcacc);
}

// ---------------------------------------------------------------------------
extern "C" void kernel_launch(void* const* d_in, const int* in_sizes, int n_in,
                              void* d_out, int out_size, void* d_ws, size_t ws_size,
                              hipStream_t stream) {
  const float* x   = (const float*)d_in[0];
  const float* Wq  = (const float*)d_in[1];
  const float* bq  = (const float*)d_in[2];
  const float* Wk  = (const float*)d_in[3];
  const float* bk  = (const float*)d_in[4];
  const float* Wv  = (const float*)d_in[5];
  const float* bv  = (const float*)d_in[6];
  const float* Wfc = (const float*)d_in[7];
  const float* bfc = (const float*)d_in[8];
  float* out = (float*)d_out;

  char* ws = (char*)d_ws;
  _Float16* qh = (_Float16*)(ws);                       // 4 MB
  _Float16* kh = (_Float16*)(ws + (4u << 20));          // 4 MB
  _Float16* vT = (_Float16*)(ws + (8u << 20));          // 4 MB (v transposed [B][D][S])

  qkv_wmma_kernel<<<256, 256, 0, stream>>>(x, Wq, bq, Wk, bk, Wv, bv, qh, kh, vT);
  init_out_kernel<<<1, 32, 0, stream>>>(out, bfc);
  attn_fused_kernel<<<dim3(8, B_), 256, 0, stream>>>(qh, kh, vT, Wfc, out);
}
